// AttentionBlock_76931454206433
// MI455X (gfx1250) — matrix operand order repaired and checked by reading, not compiled, not run on hardware
//
#include <hip/hip_runtime.h>

// ---------------- problem constants ----------------
#define B_   4
#define C_   512
#define S_   4096          // 64*64 spatial
#define G_   32            // groups
#define GCH  16            // channels per group
#define QSCALE 0.04419417382415922f   // 1/sqrt(512)

typedef __attribute__((ext_vector_type(16))) _Float16 v16h;
typedef __attribute__((ext_vector_type(8)))  _Float16 v8h;
typedef __attribute__((ext_vector_type(8)))  float    v8f;

// 16-byte int vector type matching the async-LDS builtin's parameter type
typedef int i128_t __attribute__((vector_size(16)));
#define AS_GLOBAL __attribute__((address_space(1)))
#define AS_LDS    __attribute__((address_space(3)))

#define LDS_PAD 40   // padded row (f16 elems) for Kstep=32 tiles: keeps 16B align, spreads banks

// CDNA5 async global->LDS path (ASYNCcnt-tracked); guarded so either toolchain compiles.
#if __has_builtin(__builtin_amdgcn_global_load_async_to_lds_b128) && __has_builtin(__builtin_amdgcn_s_wait_asynccnt)
#define ASYNC_LDS 1
#else
#define ASYNC_LDS 0
#endif

// wait until at most PEND async ops remain in flight (immediate operand)
template<int PEND>
__device__ __forceinline__ void async_wait_pend() {
#if ASYNC_LDS
    __builtin_amdgcn_s_wait_asynccnt(PEND);
#endif
}

// =====================================================================
// tile movers: 128 rows x 32 cols f16 (row-major global, leading dim ld)
// issues 2 async-LDS instructions per wave (all lanes each move 16B)
// =====================================================================
__device__ __forceinline__ void load_tile16_128(_Float16* lds, const _Float16* g, int ld) {
    int t = threadIdx.x;
#pragma unroll
    for (int p = 0; p < 2; ++p) {
        int idx = t + p * 256;          // 0..511
        int row = idx >> 2;             // 0..127
        int cg  = (idx & 3) * 8;        // 0,8,16,24
#if ASYNC_LDS
        __builtin_amdgcn_global_load_async_to_lds_b128(
            (AS_GLOBAL i128_t*)(g + (size_t)row * ld + cg),
            (AS_LDS    i128_t*)(lds + row * LDS_PAD + cg),
            0, 0);
#else
        *(v8h*)(lds + row * LDS_PAD + cg) = *(const v8h*)(g + (size_t)row * ld + cg);
#endif
    }
}

// 128 x 32 tile from f32 global (softmaxed scores), converted to f16 in LDS
// synchronous path: ordered by the DScnt wait the compiler emits at barriers
__device__ __forceinline__ void load_tile32_128(_Float16* lds, const float* g, int ld) {
    int t = threadIdx.x;
#pragma unroll
    for (int p = 0; p < 2; ++p) {
        int idx = t + p * 256;
        int row = idx >> 2;
        int cg  = (idx & 3) * 8;
        const float* src = g + (size_t)row * ld + cg;
        _Float16* d = lds + row * LDS_PAD + cg;
#pragma unroll
        for (int i = 0; i < 8; ++i) d[i] = (_Float16)src[i];
    }
}

// =====================================================================
// WMMA fragments (16x16x32 f16, wave32 layouts per CDNA5 ISA 7.12.2)
// =====================================================================
union FragU { v16h v; v8h h[2]; };

// A fragment from MxK tile: lane L<16 -> row L, K {0..7,16..23}; lane L+16 -> K {8..15,24..31}
__device__ __forceinline__ v16h frag_A(const _Float16* lds, int tm, int lane) {
    int row = tm * 16 + (lane & 15);
    int kb  = (lane & 16) ? 8 : 0;
    const _Float16* p = lds + row * LDS_PAD + kb;
    FragU f;
    f.h[0] = *(const v8h*)(p);
    f.h[1] = *(const v8h*)(p + 16);
    return f.v;
}

// B fragment from NxK tile: lane L<16 -> col L, K {0..7,16..23}; lane L+16 -> col L, K {8..15,24..31}
__device__ __forceinline__ v16h frag_B(const _Float16* lds, int tn, int lane) {
    int row = tn * 16 + (lane & 15);
    int kb  = (lane & 16) ? 8 : 0;
    const _Float16* p = lds + row * LDS_PAD + kb;
    FragU f;
    f.h[0] = *(const v8h*)(p);
    f.h[1] = *(const v8h*)(p + 16);
    return f.v;
}

__device__ __forceinline__ v8f wmma_f16(v16h a, v16h b, v8f c) {
    return __builtin_amdgcn_wmma_f32_16x16x32_f16(false, a, false, b, (short)0, c, false, false);
}

// one K-step: wave computes 2(M) x 4(N) subtiles = 8 WMMAs from 6 fragment loads
__device__ __forceinline__ void mma_step(const _Float16* As, const _Float16* Bs,
                                         int wrow, int wcol, int lane, v8f acc[8]) {
    v16h a0 = frag_A(As, wrow * 2 + 0, lane);
    v16h a1 = frag_A(As, wrow * 2 + 1, lane);
#pragma unroll
    for (int j = 0; j < 4; ++j) {
        v16h b = frag_B(Bs, wcol * 4 + j, lane);
        acc[j]     = wmma_f16(a0, b, acc[j]);
        acc[4 + j] = wmma_f16(a1, b, acc[4 + j]);
    }
}

// =====================================================================
// small prep kernels
// =====================================================================
__global__ void f32_to_f16_kernel(const float* __restrict__ src, _Float16* __restrict__ dst, int n) {
    int i = blockIdx.x * blockDim.x + threadIdx.x;
    if (i < n) dst[i] = (_Float16)src[i];
}

__global__ __launch_bounds__(256) void gn_stats_kernel(const float* __restrict__ x,
                                                       float* __restrict__ stats) {
    int bg = blockIdx.x;                       // 0..127 = (b,group)
    const float* p = x + (size_t)bg * GCH * S_;
    const int n = GCH * S_;                    // 65536 contiguous
    float s = 0.f, ss = 0.f;
    for (int i = threadIdx.x; i < n; i += 256) {
        float v = p[i];
        s += v; ss += v * v;
    }
    __shared__ float r1[256], r2[256];
    r1[threadIdx.x] = s; r2[threadIdx.x] = ss;
    __syncthreads();
    for (int st = 128; st > 0; st >>= 1) {
        if (threadIdx.x < st) { r1[threadIdx.x] += r1[threadIdx.x + st]; r2[threadIdx.x] += r2[threadIdx.x + st]; }
        __syncthreads();
    }
    if (threadIdx.x == 0) {
        float mu  = r1[0] / (float)n;
        float var = r2[0] / (float)n - mu * mu;
        stats[bg * 2 + 0] = mu;
        stats[bg * 2 + 1] = rsqrtf(var + 1e-5f);
    }
}

// normalize + affine, write hT[b][s][c] (f16, NxK-natural for the qkv GEMM)
__global__ __launch_bounds__(256) void gn_apply_kernel(const float* __restrict__ x,
                                                       const float* __restrict__ gw,
                                                       const float* __restrict__ gb,
                                                       const float* __restrict__ stats,
                                                       _Float16* __restrict__ hT) {
    size_t idx = (size_t)blockIdx.x * blockDim.x + threadIdx.x;   // over B*C*S
    int s = (int)(idx % S_);
    size_t r = idx / S_;
    int c = (int)(r % C_);
    int b = (int)(r / C_);
    int bg = b * G_ + (c >> 4);
    float mu = stats[bg * 2 + 0];
    float rs = stats[bg * 2 + 1];
    float v = (x[idx] - mu) * rs * gw[c] + gb[c];
    hT[((size_t)b * S_ + s) * C_ + c] = (_Float16)v;
}

// =====================================================================
// GEMM 1: q/k/v = W(512x512) x h + bias.  grid.z = b*3+{q,k,v}
//   128x128 tile, double-buffered async LDS fill
// =====================================================================
__global__ __launch_bounds__(256) void gemm_qkv_kernel(
    const _Float16* __restrict__ Wq, const _Float16* __restrict__ Wk, const _Float16* __restrict__ Wv,
    const float* __restrict__ bq, const float* __restrict__ bk, const float* __restrict__ bv,
    const _Float16* __restrict__ hT,
    _Float16* __restrict__ qT, _Float16* __restrict__ kT, _Float16* __restrict__ vO) {
    __shared__ _Float16 As[2][128 * LDS_PAD];
    __shared__ _Float16 Bs[2][128 * LDS_PAD];
    int z = blockIdx.z;
    int b = z / 3, which = z % 3;
    const _Float16* W    = (which == 0) ? Wq : (which == 1) ? Wk : Wv;
    const float*    bias = (which == 0) ? bq : (which == 1) ? bk : bv;
    const _Float16* Bsrc = hT + (size_t)b * S_ * C_;

    int m0 = blockIdx.y * 128;    // output channel o
    int n0 = blockIdx.x * 128;    // spatial s
    int lane = threadIdx.x & 31;
    int wave = threadIdx.x >> 5;
    int wrow = wave & 3;          // -> tm = wrow*2 + {0,1}
    int wcol = wave >> 2;         // -> tn = wcol*4 + {0..3}

    auto loadAB = [&](int buf, int k0) {
        load_tile16_128(As[buf], W + (size_t)m0 * C_ + k0, C_);
        load_tile16_128(Bs[buf], Bsrc + (size_t)n0 * C_ + k0, C_);
    };
    v8f acc[8] = {};
    const int NS = C_ / 32;
    loadAB(0, 0);
    for (int ks = 0; ks < NS; ++ks) {
        int cur = ks & 1;
        if (ks + 1 < NS) { loadAB(cur ^ 1, (ks + 1) * 32); async_wait_pend<4>(); }
        else             { async_wait_pend<0>(); }
        __syncthreads();
        mma_step(As[cur], Bs[cur], wrow, wcol, lane, acc);
        __syncthreads();
    }
    int rb = (lane & 16) ? 8 : 0;
    int nc = lane & 15;
#pragma unroll
    for (int t = 0; t < 2; ++t) {
        int mb = m0 + (wrow * 2 + t) * 16 + rb;   // first of 8 consecutive output channels
#pragma unroll
        for (int j = 0; j < 4; ++j) {
            v8f a = acc[t * 4 + j];
#pragma unroll
            for (int i = 0; i < 8; ++i) a[i] += bias[mb + i];
            int sg = n0 + (wcol * 4 + j) * 16 + nc;
            if (which < 2) {
                _Float16* dst = ((which == 0) ? qT : kT) + (size_t)b * S_ * C_;
                float sc = (which == 0) ? QSCALE : 1.0f;
                v8h o;
#pragma unroll
                for (int i = 0; i < 8; ++i) o[i] = (_Float16)(a[i] * sc);
                *(v8h*)(dst + (size_t)sg * C_ + mb) = o;          // transposed, contiguous per lane
            } else {
                _Float16* dst = vO + (size_t)b * C_ * S_;
#pragma unroll
                for (int i = 0; i < 8; ++i) dst[(size_t)(mb + i) * S_ + sg] = (_Float16)a[i];
            }
        }
    }
}

// =====================================================================
// GEMM 2: scores[i][j] = sum_c qT[i][c]*kT[j][c]  (scale folded into q)
//   double-buffered; f32 out staged via 128x64 LDS slab (x2) -> coalesced
// =====================================================================
__global__ __launch_bounds__(256) void gemm_scores_kernel(
    const _Float16* __restrict__ qTb, const _Float16* __restrict__ kTb,
    float* __restrict__ scores) {
    __shared__ _Float16 As[2][128 * LDS_PAD];
    __shared__ _Float16 Bs[2][128 * LDS_PAD];
    __shared__ float Cs[128][65];

    int m0 = blockIdx.y * 128;    // i
    int n0 = blockIdx.x * 128;    // j
    int lane = threadIdx.x & 31;
    int wave = threadIdx.x >> 5;
    int wrow = wave & 3;
    int wcol = wave >> 2;

    auto loadAB = [&](int buf, int k0) {
        load_tile16_128(As[buf], qTb + (size_t)m0 * C_ + k0, C_);
        load_tile16_128(Bs[buf], kTb + (size_t)n0 * C_ + k0, C_);
    };
    v8f acc[8] = {};
    const int NS = C_ / 32;
    loadAB(0, 0);
    for (int ks = 0; ks < NS; ++ks) {
        int cur = ks & 1;
        if (ks + 1 < NS) { loadAB(cur ^ 1, (ks + 1) * 32); async_wait_pend<4>(); }
        else             { async_wait_pend<0>(); }
        __syncthreads();
        mma_step(As[cur], Bs[cur], wrow, wcol, lane, acc);
        __syncthreads();
    }
    int rb = (lane & 16) ? 8 : 0;
    int nc = lane & 15;
#pragma unroll
    for (int pass = 0; pass < 2; ++pass) {
        __syncthreads();
        if (wcol == pass) {
#pragma unroll
            for (int t = 0; t < 2; ++t) {
                int mr = (wrow * 2 + t) * 16 + rb;
#pragma unroll
                for (int j = 0; j < 4; ++j) {
                    int ncl = j * 16 + nc;                 // 0..63 within slab
#pragma unroll
                    for (int i = 0; i < 8; ++i) Cs[mr + i][ncl] = acc[t * 4 + j][i];
                }
            }
        }
        __syncthreads();
        int row = threadIdx.x >> 1;
        int off = (threadIdx.x & 1) * 32;
        float* gp = scores + (size_t)(m0 + row) * S_ + n0 + pass * 64 + off;
#pragma unroll
        for (int j = 0; j < 32; ++j) gp[j] = Cs[row][off + j];
    }
}

// row softmax in place on f32 scores
__global__ __launch_bounds__(256) void softmax_kernel(float* __restrict__ scores) {
    float* p = scores + (size_t)blockIdx.x * S_;
    __shared__ float red[256];
    int t = threadIdx.x;
    float mx = -1e30f;
    for (int j = t; j < S_; j += 256) mx = fmaxf(mx, p[j]);
    red[t] = mx; __syncthreads();
    for (int s = 128; s > 0; s >>= 1) { if (t < s) red[t] = fmaxf(red[t], red[t + s]); __syncthreads(); }
    mx = red[0]; __syncthreads();
    float sum = 0.f;
    for (int j = t; j < S_; j += 256) { float e = __expf(p[j] - mx); p[j] = e; sum += e; }
    red[t] = sum; __syncthreads();
    for (int s = 128; s > 0; s >>= 1) { if (t < s) red[t] += red[t + s]; __syncthreads(); }
    float inv = 1.0f / red[0];
    for (int j = t; j < S_; j += 256) p[j] *= inv;
}

// =====================================================================
// GEMM 3: attnOut[c][i] = sum_j v[c][j] * P[i][j]
//   A = v (async, 2 ops/wave), B = P f32->f16 convert (sync) ; K = 4096
//   store aT[i][c] (contig per lane) -> NxK operand for proj GEMM
// =====================================================================
__global__ __launch_bounds__(256) void gemm_attnv_kernel(
    const _Float16* __restrict__ vb, const float* __restrict__ P,
    _Float16* __restrict__ aTb) {
    __shared__ _Float16 As[2][128 * LDS_PAD];
    __shared__ _Float16 Bs[2][128 * LDS_PAD];

    int m0 = blockIdx.y * 128;    // c
    int n0 = blockIdx.x * 128;    // i
    int lane = threadIdx.x & 31;
    int wave = threadIdx.x >> 5;
    int wrow = wave & 3;
    int wcol = wave >> 2;

    auto loadAB = [&](int buf, int k0) {
        load_tile16_128(As[buf], vb + (size_t)m0 * S_ + k0, S_);
        load_tile32_128(Bs[buf], P + (size_t)n0 * S_ + k0, S_);
    };
    v8f acc[8] = {};
    const int NS = S_ / 32;
    loadAB(0, 0);
    for (int ks = 0; ks < NS; ++ks) {
        int cur = ks & 1;
        if (ks + 1 < NS) { loadAB(cur ^ 1, (ks + 1) * 32); async_wait_pend<2>(); }
        else             { async_wait_pend<0>(); }
        __syncthreads();
        mma_step(As[cur], Bs[cur], wrow, wcol, lane, acc);
        __syncthreads();
    }
    int rb = (lane & 16) ? 8 : 0;
    int nc = lane & 15;
#pragma unroll
    for (int t = 0; t < 2; ++t) {
        int mb = m0 + (wrow * 2 + t) * 16 + rb;
#pragma unroll
        for (int j = 0; j < 4; ++j) {
            int ig = n0 + (wcol * 4 + j) * 16 + nc;
            v8h o;
#pragma unroll
            for (int i = 0; i < 8; ++i) o[i] = (_Float16)acc[t * 4 + j][i];
            *(v8h*)(aTb + (size_t)ig * C_ + mb) = o;
        }
    }
}

// =====================================================================
// GEMM 4: out[b][o][s] = x[b][o][s] + bp[o] + sum_c Wp[o][c]*aT[s][c]
//   double-buffered; staged through LDS slab for coalesced residual+store
// =====================================================================
__global__ __launch_bounds__(256) void gemm_proj_kernel(
    const _Float16* __restrict__ Wp, const float* __restrict__ bp,
    const _Float16* __restrict__ aT, const float* __restrict__ x,
    float* __restrict__ out) {
    __shared__ _Float16 As[2][128 * LDS_PAD];
    __shared__ _Float16 Bs[2][128 * LDS_PAD];
    __shared__ float Cs[128][65];

    int b  = blockIdx.z;
    int m0 = blockIdx.y * 128;    // o
    int n0 = blockIdx.x * 128;    // s
    const _Float16* Bsrc = aT + (size_t)b * S_ * C_;
    int lane = threadIdx.x & 31;
    int wave = threadIdx.x >> 5;
    int wrow = wave & 3;
    int wcol = wave >> 2;

    auto loadAB = [&](int buf, int k0) {
        load_tile16_128(As[buf], Wp + (size_t)m0 * C_ + k0, C_);
        load_tile16_128(Bs[buf], Bsrc + (size_t)n0 * C_ + k0, C_);
    };
    v8f acc[8] = {};
    const int NS = C_ / 32;
    loadAB(0, 0);
    for (int ks = 0; ks < NS; ++ks) {
        int cur = ks & 1;
        if (ks + 1 < NS) { loadAB(cur ^ 1, (ks + 1) * 32); async_wait_pend<4>(); }
        else             { async_wait_pend<0>(); }
        __syncthreads();
        mma_step(As[cur], Bs[cur], wrow, wcol, lane, acc);
        __syncthreads();
    }
    int rb = (lane & 16) ? 8 : 0;
    int nc = lane & 15;
#pragma unroll
    for (int pass = 0; pass < 2; ++pass) {
        __syncthreads();
        if (wcol == pass) {
#pragma unroll
            for (int t = 0; t < 2; ++t) {
                int mr = (wrow * 2 + t) * 16 + rb;
#pragma unroll
                for (int j = 0; j < 4; ++j) {
                    int ncl = j * 16 + nc;
#pragma unroll
                    for (int i = 0; i < 8; ++i) Cs[mr + i][ncl] = acc[t * 4 + j][i];
                }
            }
        }
        __syncthreads();
        int row = threadIdx.x >> 1;
        int off = (threadIdx.x & 1) * 32;
        float bb = bp[m0 + row];
        size_t gidx = ((size_t)b * C_ + m0 + row) * S_ + n0 + pass * 64 + off;
#pragma unroll
        for (int j = 0; j < 32; ++j) out[gidx + j] = Cs[row][off + j] + bb + x[gidx + j];
    }
}

// =====================================================================
// host launch
// =====================================================================
extern "C" void kernel_launch(void* const* d_in, const int* in_sizes, int n_in,
                              void* d_out, int out_size, void* d_ws, size_t ws_size,
                              hipStream_t stream) {
    const float* x   = (const float*)d_in[0];
    const float* gnw = (const float*)d_in[1];
    const float* gnb = (const float*)d_in[2];
    const float* wq  = (const float*)d_in[3];
    const float* bq  = (const float*)d_in[4];
    const float* wk  = (const float*)d_in[5];
    const float* bk  = (const float*)d_in[6];
    const float* wv  = (const float*)d_in[7];
    const float* bv  = (const float*)d_in[8];
    const float* wp  = (const float*)d_in[9];
    const float* bp  = (const float*)d_in[10];
    float* out = (float*)d_out;

    // ---- workspace carve-up (all 256B aligned) ----
    char* ws = (char*)d_ws;
    size_t off = 0;
    auto alloc = [&](size_t bytes) -> void* {
        void* p = ws + off;
        off += (bytes + 255) & ~(size_t)255;
        return p;
    };
    const size_t WELEMS = (size_t)C_ * C_;
    _Float16* Wq16 = (_Float16*)alloc(WELEMS * 2);
    _Float16* Wk16 = (_Float16*)alloc(WELEMS * 2);
    _Float16* Wv16 = (_Float16*)alloc(WELEMS * 2);
    _Float16* Wp16 = (_Float16*)alloc(WELEMS * 2);
    float*    stats = (float*)alloc((size_t)B_ * G_ * 2 * 4);
    const size_t TEN = (size_t)B_ * S_ * C_;          // 8.4M elems
    _Float16* hT = (_Float16*)alloc(TEN * 2);
    _Float16* qT = (_Float16*)alloc(TEN * 2);
    _Float16* kT = (_Float16*)alloc(TEN * 2);
    _Float16* vB = (_Float16*)alloc(TEN * 2);
    _Float16* aT = (_Float16*)alloc(TEN * 2);
    float* scores = (float*)alloc((size_t)S_ * S_ * 4);   // per-batch reuse

    // ---- prep ----
    int wn = (int)WELEMS;
    f32_to_f16_kernel<<<wn / 256, 256, 0, stream>>>(wq, Wq16, wn);
    f32_to_f16_kernel<<<wn / 256, 256, 0, stream>>>(wk, Wk16, wn);
    f32_to_f16_kernel<<<wn / 256, 256, 0, stream>>>(wv, Wv16, wn);
    f32_to_f16_kernel<<<wn / 256, 256, 0, stream>>>(wp, Wp16, wn);

    gn_stats_kernel<<<B_ * G_, 256, 0, stream>>>(x, stats);
    gn_apply_kernel<<<(int)(((size_t)B_ * C_ * S_) / 256), 256, 0, stream>>>(x, gnw, gnb, stats, hT);

    // ---- q,k,v GEMMs: all batches + all three projections in one grid ----
    gemm_qkv_kernel<<<dim3(S_ / 128, C_ / 128, B_ * 3), 256, 0, stream>>>(
        Wq16, Wk16, Wv16, bq, bk, bv, hT, qT, kT, vB);

    // ---- attention core, per batch (scores buffer reused) ----
    for (int b = 0; b < B_; ++b) {
        const _Float16* qTb = qT + (size_t)b * S_ * C_;
        const _Float16* kTb = kT + (size_t)b * S_ * C_;
        const _Float16* vb  = vB + (size_t)b * C_ * S_;
        _Float16*       aTb = aT + (size_t)b * S_ * C_;
        gemm_scores_kernel<<<dim3(S_ / 128, S_ / 128), 256, 0, stream>>>(qTb, kTb, scores);
        softmax_kernel<<<S_, 256, 0, stream>>>(scores);
        gemm_attnv_kernel<<<dim3(S_ / 128, C_ / 128), 256, 0, stream>>>(vb, scores, aTb);
    }

    // ---- projection + bias + residual ----
    gemm_proj_kernel<<<dim3(S_ / 128, C_ / 128, B_), 256, 0, stream>>>(Wp16, bp, aT, x, out);
}
